// VectorQuantizer_36833639530977
// MI455X (gfx1250) — matrix operand order repaired and checked by reading, not compiled, not run on hardware
//
#include <hip/hip_runtime.h>
#include <hip/hip_bf16.h>

typedef __attribute__((ext_vector_type(16))) __bf16 v16bf;
typedef __attribute__((ext_vector_type(8)))  __bf16 v8bf;
typedef __attribute__((ext_vector_type(8)))  float  v8f;
typedef __attribute__((ext_vector_type(4)))  float  v4f;

#define HW        4096                 // 64*64
#define LOSS_OFF  33554432u
#define IDX_OFF   33554434u

// dynamic LDS layout
#define SH_A_BYTES 69632               // 256 rows * 136 bf16 * 2B (padded, 16B-aligned rows)
#define SH_B_BYTES 24576               // ring-of-3 8KB B group tiles
#define SH_MN_OFF  (SH_A_BYTES + SH_B_BYTES)        // 94208
#define SH_ID_OFF  (SH_MN_OFF + 1024)               // 95232
#define SH_RED_OFF (SH_ID_OFF + 1024)               // 96256
#define SMEM_BYTES (SH_RED_OFF + 1024)              // 97280

// CDNA5 async global->LDS copies (gated: fall back to reg-prefetch if absent)
#if defined(__has_builtin)
#  if __has_builtin(__builtin_amdgcn_global_load_async_to_lds_b128) && \
      __has_builtin(__builtin_amdgcn_s_wait_asynccnt)
#    define HAVE_ASYNC_LDS 1
#  endif
#endif

#ifdef HAVE_ASYNC_LDS
// builtin prototype uses gcc-style v4i32 pointers: (v4i as1*, v4i as3*, imm off, imm cpol)
typedef int v4i_ __attribute__((__vector_size__(16)));
typedef __attribute__((address_space(1))) v4i_* as1_v4i_p;
typedef __attribute__((address_space(3))) v4i_* as3_v4i_p;
__device__ __forceinline__ void async_cp16(const void* gsrc, void* ldst) {
    // GLOBAL_LOAD_ASYNC_TO_LDS_B128: 16B per lane, tracked by ASYNCcnt
    __builtin_amdgcn_global_load_async_to_lds_b128(
        (as1_v4i_p)(uintptr_t)gsrc,
        (as3_v4i_p)(uintptr_t)(uint32_t)(uintptr_t)ldst,   // low 32 bits = LDS offset
        0, 0);
}
#endif

// ws layout:
//   [0, 512K)       : B matrix, bf16, scaled by -2, WMMA-B-swizzled
//   [512K, 512K+4K) : esq[k] = ||codebook_k||^2 (fp32)
//   [512K+4K, +4K)  : per-block loss partials (fp32, 512 entries)

// codebook -> bf16 * (-2), B tile in this kernel's contiguous-k order, same as A (32x16 K x N):
// lane L: n = L%16 ; element j: K = (L/16)*16 + j
__global__ __launch_bounds__(256)
void vq_prep_b(const float* __restrict__ cb, __bf16* __restrict__ wsb) {
    int tid = blockIdx.x * 256 + threadIdx.x;   // < 262144
    int j  = tid & 15;
    int L  = (tid >> 4) & 31;
    int kc = (tid >> 9) & 7;
    int g  = tid >> 12;
    int code = g * 16 + (L & 15);
    int ch   = kc * 32 + ((L >> 4) << 4) + j;
    wsb[tid] = (__bf16)(-2.0f * cb[code * 256 + ch]);
}

__global__ __launch_bounds__(256)
void vq_prep_esq(const float* __restrict__ cb, float* __restrict__ esq) {
    int k = blockIdx.x * 256 + threadIdx.x;     // < 1024
    float s = 0.f;
    for (int c = 0; c < 256; ++c) { float v = cb[k * 256 + c]; s += v * v; }
    esq[k] = s;
}

// main: 512 blocks, each 256 rows x all 1024 codes; 8 waves x 32 rows (2 M-tiles)
__global__ __launch_bounds__(256)
void vq_main(const float* __restrict__ z, const float* __restrict__ cb,
             const __bf16* __restrict__ wsb, const float* __restrict__ esq,
             float* __restrict__ out, float* __restrict__ partial) {
    extern __shared__ char smem[];
    __bf16* shA   = (__bf16*)smem;
    __bf16* shB   = (__bf16*)(smem + SH_A_BYTES);
    float*  shMn  = (float*)(smem + SH_MN_OFF);
    int*    shId  = (int*)(smem + SH_ID_OFF);
    float*  shRed = (float*)(smem + SH_RED_OFF);

    const int t    = threadIdx.x;
    const int lane = t & 31;
    const int wave = t >> 5;
    const int blk  = blockIdx.x;
    const int b    = blk >> 4;            // batch
    const int r0   = (blk & 15) * 256;    // spatial row offset within batch

    union AF { v16bf v; v8bf h[2]; };
    AF afrag[2][8];
    const int m16  = lane & 15;
    const int sel8 = ((lane >> 4) & 1) * 16;  // A layout: lanes>=16 shifted +16 in K

    // ---- stage z (fp32->bf16) through LDS, 2 passes of 128 channels;
    //      thread t owns row t, so zacc ends as full ||z_row||^2
    float zacc = 0.f;
    for (int p = 0; p < 2; ++p) {
        for (int i = 0; i < 128; ++i) {
            float v = z[(size_t)(b * 256 + p * 128 + i) * HW + r0 + t];
            zacc += v * v;
            shA[t * 136 + i] = (__bf16)v;
        }
        __syncthreads();
#pragma unroll
        for (int tile = 0; tile < 2; ++tile) {
            const int arow = wave * 32 + tile * 16 + m16;
#pragma unroll
            for (int fl = 0; fl < 4; ++fl) {
                const __bf16* ap = &shA[arow * 136 + fl * 32 + sel8];
                afrag[tile][p * 4 + fl].h[0] = *(const v8bf*)ap;
                afrag[tile][p * 4 + fl].h[1] = *(const v8bf*)(ap + 8);
            }
        }
        __syncthreads();
    }

    // ---- B staging: ring of 3 LDS buffers, data for group g+2 staged at iter g.
    //      coop: thread t copies 32B of group tile (wsb is contiguous per group).
#ifdef HAVE_ASYNC_LDS
    {
#pragma unroll
        for (int g0 = 0; g0 < 2; ++g0) {
            const char* gp = (const char*)wsb + ((size_t)g0 * 256 + t) * 32;
            __bf16* d = shB + g0 * 4096 + (t >> 5) * 512 + (t & 31) * 8;
            async_cp16(gp,      d);
            async_cp16(gp + 16, d + 256);
        }
        __builtin_amdgcn_s_wait_asynccnt(0);
        __syncthreads();
    }
#else
    AF pre;
    {
#pragma unroll
        for (int g0 = 0; g0 < 2; ++g0) {
            pre.v = *(const v16bf*)(wsb + ((size_t)g0 * 256 + t) * 16);
            __bf16* d = shB + g0 * 4096 + (t >> 5) * 512 + (t & 31) * 8;
            *(v8bf*)d = pre.h[0];
            *(v8bf*)(d + 256) = pre.h[1];
        }
        __syncthreads();
    }
#endif

    float minv[2][8];
    int   mini[2][8];
#pragma unroll
    for (int tile = 0; tile < 2; ++tile)
#pragma unroll
        for (int v = 0; v < 8; ++v) { minv[tile][v] = 3.4e38f; mini[tile][v] = 0; }

    float ecur = esq[m16];
    for (int g = 0; g < 64; ++g) {
        const int cur = g % 3;
        float enext = 0.f;
        if (g + 1 < 64) enext = esq[(g + 1) * 16 + m16];
#ifndef HAVE_ASYNC_LDS
        if (g + 2 < 64)   // prefetch group g+2 B into registers while WMMAs run
            pre.v = *(const v16bf*)(wsb + ((size_t)(g + 2) * 256 + t) * 16);
#endif
        const __bf16* Bb = shB + cur * 4096;
        v8f acc0, acc1;
#pragma unroll
        for (int v = 0; v < 8; ++v) { acc0[v] = ecur; acc1[v] = ecur; }  // D = ||e||^2 - 2 z.e
#pragma unroll
        for (int f = 0; f < 8; ++f) {
            AF bf;
            bf.h[0] = *(const v8bf*)(Bb + f * 512 + lane * 8);
            bf.h[1] = *(const v8bf*)(Bb + f * 512 + 256 + lane * 8);
            acc0 = __builtin_amdgcn_wmma_f32_16x16x32_bf16(
                false, afrag[0][f].v, false, bf.v, (short)0, acc0, false, false);
            acc1 = __builtin_amdgcn_wmma_f32_16x16x32_bf16(
                false, afrag[1][f].v, false, bf.v, (short)0, acc1, false, false);
        }
        const int code = g * 16 + m16;
#pragma unroll
        for (int v = 0; v < 8; ++v) {
            if (acc0[v] < minv[0][v]) { minv[0][v] = acc0[v]; mini[0][v] = code; }
            if (acc1[v] < minv[1][v]) { minv[1][v] = acc1[v]; mini[1][v] = code; }
        }
#ifdef HAVE_ASYNC_LDS
        if (g + 1 < 64) __builtin_amdgcn_s_wait_asynccnt(0);  // group g+1 copy done
        __syncthreads();   // all waves: compute(g) done (frees buf (g+2)%3), g+1 visible
        if (g + 2 < 64) {
            const char* gp = (const char*)wsb + ((size_t)(g + 2) * 256 + t) * 32;
            __bf16* d = shB + ((g + 2) % 3) * 4096 + (t >> 5) * 512 + (t & 31) * 8;
            async_cp16(gp,      d);
            async_cp16(gp + 16, d + 256);
        }
#else
        __syncthreads();   // all waves: compute(g) done -> buf (g+2)%3 free; g+1 visible
        if (g + 2 < 64) {
            __bf16* d = shB + ((g + 2) % 3) * 4096 + (t >> 5) * 512 + (t & 31) * 8;
            *(v8bf*)d = pre.h[0];
            *(v8bf*)(d + 256) = pre.h[1];
        }
#endif
        ecur = enext;
    }

    // ---- min over the 16 lanes of each half (ties -> lower index)
#pragma unroll
    for (int off = 8; off >= 1; off >>= 1) {
#pragma unroll
        for (int tile = 0; tile < 2; ++tile)
#pragma unroll
            for (int v = 0; v < 8; ++v) {
                float ov = __shfl_xor(minv[tile][v], off, 32);
                int   oi = __shfl_xor(mini[tile][v], off, 32);
                if (ov < minv[tile][v] || (ov == minv[tile][v] && oi < mini[tile][v])) {
                    minv[tile][v] = ov; mini[tile][v] = oi;
                }
            }
    }
    if (m16 == 0) {
#pragma unroll
        for (int tile = 0; tile < 2; ++tile) {
            int mbase = wave * 32 + tile * 16 + ((lane >> 4) << 3);
#pragma unroll
            for (int v = 0; v < 8; ++v) {
                shMn[mbase + v] = minv[tile][v];
                shId[mbase + v] = mini[tile][v];
            }
        }
    }
    __syncthreads();

    // ---- z_q writeback (exact fp32 codebook rows, BCHW, coalesced) + indices
    const int myidx = shId[t];
    out[(size_t)IDX_OFF + b * HW + r0 + t] = (float)myidx;
    for (int c = 0; c < 256; c += 4) {
        v4f rowv = *(const v4f*)(cb + myidx * 256 + c);
#pragma unroll
        for (int q = 0; q < 4; ++q)
            out[(size_t)(b * 256 + c + q) * HW + r0 + t] = rowv[q];
    }

    // ---- block loss partial: sum_n (||z_n||^2 + min score_n), fixed-order tree
    shRed[t] = zacc + shMn[t];
    __syncthreads();
    for (int s = 128; s > 0; s >>= 1) {
        if (t < s) shRed[t] += shRed[t + s];
        __syncthreads();
    }
    if (t == 0) partial[blk] = shRed[0];
}

__global__ __launch_bounds__(256)
void vq_finalize(const float* __restrict__ partial, float* __restrict__ out) {
    __shared__ float sh[256];
    int t = threadIdx.x;
    sh[t] = partial[t] + partial[t + 256];   // 512 partials
    __syncthreads();
    for (int k = 128; k > 0; k >>= 1) {
        if (t < k) sh[t] += sh[t + k];
        __syncthreads();
    }
    if (t == 0) {
        float cl = sh[0] * (1.0f / 33554432.0f);   // mean over B*H*W*C
        out[LOSS_OFF]     = cl;                    // codebook loss
        out[LOSS_OFF + 1] = 0.25f * cl;            // commitment loss
    }
}

extern "C" void kernel_launch(void* const* d_in, const int* in_sizes, int n_in,
                              void* d_out, int out_size, void* d_ws, size_t ws_size,
                              hipStream_t stream) {
    const float* z  = (const float*)d_in[0];   // [32,256,64,64]
    const float* cb = (const float*)d_in[1];   // [1024,256]
    float* out = (float*)d_out;
    __bf16* wsb    = (__bf16*)d_ws;                              // 512 KB
    float* esq     = (float*)((char*)d_ws + 512 * 1024);         // 4 KB
    float* partial = (float*)((char*)d_ws + 512 * 1024 + 4096);  // 2 KB used

    vq_prep_b  <<<1024, 256, 0, stream>>>(cb, wsb);
    vq_prep_esq<<<4,    256, 0, stream>>>(cb, esq);
    vq_main    <<<512,  256, SMEM_BYTES, stream>>>(z, cb, wsb, esq, out, partial);
    vq_finalize<<<1,    256, 0, stream>>>(partial, out);
}
